// SoftmaxPermutationMatrix_68762426409278
// MI455X (gfx1250) — matrix operand order repaired and checked by reading, not compiled, not run on hardware
//
#include <hip/hip_runtime.h>

// ---------------------------------------------------------------------------
// SoftmaxPermutationMatrix for MI455X (gfx1250, wave32, WMMA)
//   out[b,x,y] = softmax_y( <xhat,yhat>/tau ) * mx[b,x] * my[b,y]
// Path: fp32 -> L2-normalize -> f16 features (8MB, L2-resident) ->
//       v_wmma_f32_16x16x32_f16 two-sweep softmax (no max pass needed:
//       |logit| <= 1/tau = 5, so exp never overflows) -> one 512MB write.
// exp(x*INV_TAU) computed as v_exp_f32(x * (INV_TAU*log2e)): single fused
// constant multiply instead of two pk_mul chains.
// ---------------------------------------------------------------------------

typedef __attribute__((ext_vector_type(16))) _Float16 v16h;
typedef __attribute__((ext_vector_type(8)))  float    v8f;
typedef __attribute__((ext_vector_type(4)))  float    f4;
typedef __attribute__((ext_vector_type(4)))  _Float16 h4;

#define BB      2
#define NX      8192
#define NY      8192
#define DD      128
#define MBLOCK  64       // rows per workgroup (4 waves x 16 rows)
// (1/tau) * log2(e) = 5 * 1.4426950408889634
#define EXP2_SCALE 7.2134752044448169f

union ABChunk { v16h h; f4 f[2]; };

// ---- Kernel 1: L2-normalize rows of [rows, 128] fp32 -> f16 ----------------
// One wave per row: each lane loads float4 (32 lanes * 4 = 128), wave-reduce
// sum of squares via shfl_xor, scale + pack to f16.
__global__ __launch_bounds__(256)
void norm_rows_f16(const float* __restrict__ in, _Float16* __restrict__ out,
                   int nrows) {
  const int gw   = (blockIdx.x * blockDim.x + threadIdx.x) >> 5;
  const int lane = threadIdx.x & 31;
  if (gw >= nrows) return;

  f4 v = ((const f4*)(in + (size_t)gw * DD))[lane];
  float ss = v.x * v.x + v.y * v.y + v.z * v.z + v.w * v.w;
#pragma unroll
  for (int m = 1; m < 32; m <<= 1) ss += __shfl_xor(ss, m, 32);

  const float inv = 1.0f / fmaxf(sqrtf(ss), 1e-12f);
  h4 o;
  o.x = (_Float16)(v.x * inv);
  o.y = (_Float16)(v.y * inv);
  o.z = (_Float16)(v.z * inv);
  o.w = (_Float16)(v.w * inv);
  *(h4*)(out + (size_t)gw * DD + lane * 4) = o;
}

// ---- Kernel 2: GEMM + two-sweep softmax ------------------------------------
// Wave owns 16 rows. Sweep 1 over all Ny tiles: acc = X*Y^T via 4 WMMAs
// (K=128), s[r] += 2^(acc*EXP2_SCALE). Reduce sums across the 16-lane halves
// (rows 0-7 in lanes 0-15, rows 8-15 in lanes 16-31, matching C/D layout).
// Sweep 2 recomputes logits and writes exp * maskx/sum * masky.
__global__ __launch_bounds__(128)
void softperm_wmma(const _Float16* __restrict__ xh,
                   const _Float16* __restrict__ yh,
                   const float* __restrict__ mask_x,
                   const float* __restrict__ mask_y,
                   float* __restrict__ out) {
  const int b    = blockIdx.y;
  const int wave = threadIdx.x >> 5;
  const int lane = threadIdx.x & 31;
  const int half = lane >> 4;        // K-half selector
  const int l16  = lane & 15;        // row (A) / col (B,C,D) within tile
  const int rowBase = blockIdx.x * MBLOCK + wave * 16;

  // ---- Load A tile once: 16 rows x 128 K of f16 (32 VGPRs) ----
  // A layout per lane: elems 0-15 -> K = c*32 + half*16 + {0..15} (contig)
  const _Float16* xrow = xh + ((size_t)b * NX + rowBase + l16) * DD;
  ABChunk a[4];
#pragma unroll
  for (int c = 0; c < 4; ++c) {
    const int k0 = c * 32 + half * 16;
    a[c].f[0] = *(const f4*)(xrow + k0);
    a[c].f[1] = *(const f4*)(xrow + k0 + 8);
  }

  const _Float16* ybase = yh + (size_t)b * NY * DD;

  // ================= Sweep 1: row sums of exp(logit) =================
  float s[8];
#pragma unroll
  for (int r = 0; r < 8; ++r) s[r] = 0.0f;

  for (int n0 = 0; n0 < NY; n0 += 16) {
    const _Float16* ycol = ybase + (size_t)(n0 + l16) * DD;
    __builtin_prefetch((const void*)(ycol + 16 * DD), 0, 1);

    // B layout per lane: elems 0-15 -> K = c*32 + half*16 + {0..15} (contig)
    ABChunk bt[4];
#pragma unroll
    for (int c = 0; c < 4; ++c) {
      const int k0 = c * 32 + half * 16;
      bt[c].f[0] = *(const f4*)(ycol + k0);
      bt[c].f[1] = *(const f4*)(ycol + k0 + 8);
    }

    v8f acc = {0.f, 0.f, 0.f, 0.f, 0.f, 0.f, 0.f, 0.f};
#pragma unroll
    for (int c = 0; c < 4; ++c)
      acc = __builtin_amdgcn_wmma_f32_16x16x32_f16(
          false, a[c].h, false, bt[c].h, (short)0, acc, false, false);

#pragma unroll
    for (int r = 0; r < 8; ++r)
      s[r] += __builtin_amdgcn_exp2f(acc[r] * EXP2_SCALE);
  }

  // Reduce across lanes sharing the same rows (xor within 16-lane halves).
#pragma unroll
  for (int r = 0; r < 8; ++r) {
#pragma unroll
    for (int m = 1; m <= 8; m <<= 1) s[r] += __shfl_xor(s[r], m, 32);
  }

  float invs[8];
#pragma unroll
  for (int r = 0; r < 8; ++r) {
    const int row = rowBase + r + half * 8;   // C/D layout row for VGPR r
    invs[r] = mask_x[(size_t)b * NX + row] / s[r];
  }

  // ================= Sweep 2: recompute + write alpha =================
  float* outb = out + (size_t)b * NX * NY;
  for (int n0 = 0; n0 < NY; n0 += 16) {
    const _Float16* ycol = ybase + (size_t)(n0 + l16) * DD;
    __builtin_prefetch((const void*)(ycol + 16 * DD), 0, 1);

    ABChunk bt[4];
#pragma unroll
    for (int c = 0; c < 4; ++c) {
      const int k0 = c * 32 + half * 16;
      bt[c].f[0] = *(const f4*)(ycol + k0);
      bt[c].f[1] = *(const f4*)(ycol + k0 + 8);
    }

    v8f acc = {0.f, 0.f, 0.f, 0.f, 0.f, 0.f, 0.f, 0.f};
#pragma unroll
    for (int c = 0; c < 4; ++c)
      acc = __builtin_amdgcn_wmma_f32_16x16x32_f16(
          false, a[c].h, false, bt[c].h, (short)0, acc, false, false);

    const float my = mask_y[(size_t)b * NY + n0 + l16];
#pragma unroll
    for (int r = 0; r < 8; ++r) {
      const int row = rowBase + r + half * 8;
      outb[(size_t)row * NY + n0 + l16] =
          __builtin_amdgcn_exp2f(acc[r] * EXP2_SCALE) * invs[r] * my;
    }
  }
}

// ---------------------------------------------------------------------------
extern "C" void kernel_launch(void* const* d_in, const int* in_sizes, int n_in,
                              void* d_out, int out_size, void* d_ws,
                              size_t ws_size, hipStream_t stream) {
  const float* feat_x = (const float*)d_in[0];
  const float* feat_y = (const float*)d_in[1];
  const float* mask_x = (const float*)d_in[2];
  const float* mask_y = (const float*)d_in[3];
  float* out = (float*)d_out;

  // Workspace: f16 normalized features (4MB + 4MB)
  _Float16* xh = (_Float16*)d_ws;
  _Float16* yh = xh + (size_t)BB * NX * DD;

  // 8 rows per 256-thread block (one wave per row)
  norm_rows_f16<<<(BB * NX) / 8, 256, 0, stream>>>(feat_x, xh, BB * NX);
  norm_rows_f16<<<(BB * NY) / 8, 256, 0, stream>>>(feat_y, yh, BB * NY);

  dim3 grid(NX / MBLOCK, BB);   // 128 x 2 = 256 workgroups, 1024 waves
  softperm_wmma<<<grid, 128, 0, stream>>>(xh, yh, mask_x, mask_y, out);
}
